// TransformerBlock_5574867550255
// MI455X (gfx1250) — compile-verified, matrix operand order checked by reading, not run on hardware
//
#include <hip/hip_runtime.h>

// ---------------------------------------------------------------------------
// CDNA5 / gfx1250 transformer block forward.
// Matmuls: v_wmma_f32_16x16x32_bf16 (wave32), LDS macro-tiled GEMM with
// double-buffered GLOBAL_LOAD_ASYNC_TO_LDS_B128 staging (ASYNCcnt-tracked).
// ---------------------------------------------------------------------------

typedef __attribute__((ext_vector_type(16))) __bf16 bf16x16;
typedef __attribute__((ext_vector_type(8)))  float  f32x8;
typedef __attribute__((ext_vector_type(4)))  float  f32x4;
typedef __attribute__((ext_vector_type(4)))  int    i32x4;

#if defined(__HIP_DEVICE_COMPILE__) && defined(__has_builtin)
#if __has_builtin(__builtin_amdgcn_global_load_async_to_lds_b128) && \
    __has_builtin(__builtin_amdgcn_s_wait_asynccnt)
#define USE_ASYNC_LDS 1
#endif
#endif

#ifdef USE_ASYNC_LDS
typedef __attribute__((address_space(1))) i32x4* gas_ptr4;  // global int4*
typedef __attribute__((address_space(3))) i32x4* las_ptr4;  // LDS int4*
#endif

__device__ __forceinline__ f32x8 wmma_bf16(bf16x16 a, bf16x16 b, f32x8 c) {
  // D = A(16x32 bf16) * B(32x16 bf16) + C(16x16 f32)
  return __builtin_amdgcn_wmma_f32_16x16x32_bf16(
      /*neg_a=*/false, a, /*neg_b=*/false, b,
      /*c_mod=*/(short)0, c, /*reuse_a=*/false, /*reuse_b=*/false);
}

// K index map for 16-bit A/B fragments (wave32):
//   lane%16 = row (A) / col (B); kb = (lane/16)*8
//   element e: k = kb + e       (e in 0..7 , VGPRs 0..3)
//              k = kb + e + 8   (e in 8..15, VGPRs 4..7)
__device__ __forceinline__ int kmap(int e, int kb) {
  return kb + e + ((e >= 8) ? 8 : 0);
}

// copy 16B global -> LDS (async DMA if available, else through VGPRs)
__device__ __forceinline__ void copy16_g2l(const float* gsrc, float* ldst) {
#ifdef USE_ASYNC_LDS
  __builtin_amdgcn_global_load_async_to_lds_b128(
      (gas_ptr4)(void*)gsrc, (las_ptr4)(void*)ldst, 0, 0);
#else
  *(f32x4*)ldst = *(const f32x4*)gsrc;
#endif
}

// ---------------------------------------------------------------------------
// GEMM: C[M,N] = act(A[M,K] @ W[K,N] + bias)   (ACT: 0 = none, 1 = exact GELU)
// Block: 256 threads = 8 waves as 2(m) x 4(n); 64x128 C macro-tile per block;
// each wave owns a 32x32 sub-block = 2x2 WMMA tiles. K-step 32, double-
// buffered LDS staging (6 async b128 ops per thread per tile-pair).
// ---------------------------------------------------------------------------
template <int ACT>
__global__ __launch_bounds__(256) void gemm_bias_kernel(
    const float* __restrict__ A, const float* __restrict__ W,
    const float* __restrict__ bias, float* __restrict__ C,
    int M, int N, int K) {
  __shared__ float As[2][64][36];   // 64 rows x 32 k (pad 36: 144B rows, 16B aligned)
  __shared__ float Ws[2][32][132];  // 32 k x 128 n (pad 132: 528B rows, 16B aligned)

  const int tid = threadIdx.x;
  const int lane = tid & 31;
  const int wave = tid >> 5;
  const int wm = wave >> 2;   // 0..1
  const int wn = wave & 3;    // 0..3
  const int kb = (lane >> 4) * 8;
  const int lm = lane & 15;
  const int half = lane >> 4;

  const int gm0 = blockIdx.y * 64;
  const int gn0 = blockIdx.x * 128;

  // stage one (A 64x32, W 32x128) f32 tile pair into LDS buffer `buf`
  auto stage = [&](int buf, int k0) {
#pragma unroll
    for (int t = 0; t < 2; ++t) {           // A: 512 x 16B chunks / 256 thr
      const int c = tid + 256 * t;
      const int row = c >> 3, cj = c & 7;
      copy16_g2l(A + (size_t)(gm0 + row) * K + k0 + cj * 4, &As[buf][row][cj * 4]);
    }
#pragma unroll
    for (int t = 0; t < 4; ++t) {           // W: 1024 x 16B chunks / 256 thr
      const int c = tid + 256 * t;
      const int kr = c >> 5, cj = c & 31;
      copy16_g2l(W + (size_t)(k0 + kr) * N + gn0 + cj * 4, &Ws[buf][kr][cj * 4]);
    }
  };

  f32x8 acc[2][2] = {};
  const int nk = K / 32;
  stage(0, 0);

  for (int kt = 0; kt < nk; ++kt) {
    const int cur = kt & 1;
    if (kt + 1 < nk) {
      stage(1 - cur, (kt + 1) * 32);        // prefetch next tile pair
#ifdef USE_ASYNC_LDS
      __builtin_amdgcn_s_wait_asynccnt(6);  // 6 newer ops pending => cur done
#endif
    } else {
#ifdef USE_ASYNC_LDS
      __builtin_amdgcn_s_wait_asynccnt(0);
#endif
    }
    __syncthreads();                        // cur tile visible to all waves

    // A fragments (f32 -> bf16): two aligned b128 LDS reads per 8 elements
    bf16x16 af[2], wf[2];
#pragma unroll
    for (int mi = 0; mi < 2; ++mi) {
      const int row = wm * 32 + mi * 16 + lm;
      const f32x4 a0 = *(const f32x4*)&As[cur][row][kb];
      const f32x4 a1 = *(const f32x4*)&As[cur][row][kb + 4];
      const f32x4 a2 = *(const f32x4*)&As[cur][row][kb + 16];
      const f32x4 a3 = *(const f32x4*)&As[cur][row][kb + 20];
#pragma unroll
      for (int j = 0; j < 4; ++j) {
        af[mi][j]      = (__bf16)a0[j];
        af[mi][j + 4]  = (__bf16)a1[j];
        af[mi][j + 8]  = (__bf16)a2[j];
        af[mi][j + 12] = (__bf16)a3[j];
      }
    }
    // W fragments: 16 conflict-free b32 LDS reads (bank = 4k + n)
#pragma unroll
    for (int ni = 0; ni < 2; ++ni) {
      const int coln = wn * 32 + ni * 16 + lm;
#pragma unroll
      for (int e = 0; e < 16; ++e)
        wf[ni][e] = (__bf16)Ws[cur][kmap(e, kb)][coln];
    }
#pragma unroll
    for (int mi = 0; mi < 2; ++mi)
#pragma unroll
      for (int ni = 0; ni < 2; ++ni)
        acc[mi][ni] = wmma_bf16(af[mi], wf[ni], acc[mi][ni]);

    __syncthreads();                        // done reading cur before reuse
  }

  // epilogue: bias (+GELU), C-layout: vgpr r -> row half*8+r, lane%16 -> col
#pragma unroll
  for (int ni = 0; ni < 2; ++ni) {
    const int col = gn0 + wn * 32 + ni * 16 + lm;
    const float bv = bias[col];
#pragma unroll
    for (int mi = 0; mi < 2; ++mi) {
      const int mrow = gm0 + wm * 32 + mi * 16 + half * 8;
#pragma unroll
      for (int r = 0; r < 8; ++r) {
        float v = acc[mi][ni][r] + bv;
        if (ACT == 1) v = 0.5f * v * (1.0f + erff(v * 0.70710678118f));
        C[(size_t)(mrow + r) * N + col] = v;
      }
    }
  }
}

// ---------------------------------------------------------------------------
// Flash attention: one wave per (b, h, 16-row query tile).
// q/k/v layout: [B, S, H, DK] flattened = [B*S, 1024].
// ---------------------------------------------------------------------------
__global__ __launch_bounds__(256) void attention_kernel(
    const float* __restrict__ q, const float* __restrict__ k,
    const float* __restrict__ v, const int* __restrict__ mask,
    float* __restrict__ ctx) {
  const int S = 2048, H = 16, DK = 64, D = 1024;

  __shared__ float Pl[8][16][17];  // P tile transpose buffer (per wave)
  __shared__ float Vl[8][16][68];  // staged V tile (16 keys x 64 dims, padded)

  const int lane = threadIdx.x & 31;
  const int wave = threadIdx.x >> 5;
  const int wid = blockIdx.x * 8 + wave;   // 8192 tiles total
  const int qi = wid & 127;                // S/16
  const int h  = (wid >> 7) & (H - 1);
  const int b  = wid >> 11;
  const int kb = (lane >> 4) * 8;
  const int lm = lane & 15;
  const int half = lane >> 4;

  // Q fragments (16x64 -> two 16x32 A-frags), pre-scaled by 1/sqrt(dk)
  bf16x16 qa[2];
  {
    const int srow = qi * 16 + lm;
    const float* qp = q + (size_t)(b * S + srow) * D + h * DK;
#pragma unroll
    for (int f = 0; f < 2; ++f)
#pragma unroll
      for (int e = 0; e < 16; ++e)
        qa[f][e] = (__bf16)(qp[f * 32 + kmap(e, kb)] * 0.125f);
  }

  f32x8 o[4] = {};
  float mi[8], li[8];
#pragma unroll
  for (int r = 0; r < 8; ++r) { mi[r] = -3.0e38f; li[r] = 0.0f; }

  for (int kt = 0; kt < S / 16; ++kt) {
    // K^T fragments (B-matrix): col = key index, k-dim contiguous per lane
    const int key = kt * 16 + lm;
    const float* kp = k + (size_t)(b * S + key) * D + h * DK;
    bf16x16 kf0, kf1;
#pragma unroll
    for (int e = 0; e < 16; ++e) {
      const int kk = kmap(e, kb);
      kf0[e] = (__bf16)kp[kk];
      kf1[e] = (__bf16)kp[32 + kk];
    }
    f32x8 sc = {};
    sc = wmma_bf16(qa[0], kf0, sc);
    sc = wmma_bf16(qa[1], kf1, sc);

    // stage V tile (16 keys x 64 dims) into LDS, coalesced b128
    {
      const f32x4* vb = (const f32x4*)(v + (size_t)(b * S + kt * 16) * D + h * DK);
#pragma unroll
      for (int t = 0; t < 8; ++t) {
        const int p = lane + 32 * t;   // 256 float4 total
        const int vrow = p >> 4;       // 16 float4 per key row
        const int c4 = p & 15;
        f32x4 val = vb[(size_t)vrow * (D / 4) + c4];
        *(f32x4*)&Vl[wave][vrow][c4 * 4] = val;
      }
    }

    // online softmax (rows live in VGPR index, cols in lanes)
#pragma unroll
    for (int r = 0; r < 8; ++r) {
      const int srow = qi * 16 + half * 8 + r;
      const int mval = mask[((size_t)b * S + srow) * S + kt * 16 + lm];
      float sv = (mval == 0) ? -1.0e9f : sc[r];
      float rmax = sv;
#pragma unroll
      for (int o2 = 8; o2 > 0; o2 >>= 1) rmax = fmaxf(rmax, __shfl_xor(rmax, o2, 16));
      const float nm = fmaxf(mi[r], rmax);
      const float pexp = __expf(sv - nm);
      float rsum = pexp;
#pragma unroll
      for (int o2 = 8; o2 > 0; o2 >>= 1) rsum += __shfl_xor(rsum, o2, 16);
      const float corr = __expf(mi[r] - nm);
      li[r] = li[r] * corr + rsum;
      mi[r] = nm;
      o[0][r] *= corr; o[1][r] *= corr; o[2][r] *= corr; o[3][r] *= corr;
      Pl[wave][half * 8 + r][lm] = pexp;  // C-layout -> LDS for transpose
    }
    asm volatile("s_wait_dscnt 0" ::: "memory");  // order LDS write->read (per-wave)

    // P as A-frag (16x16 zero-padded to 16x32)
    bf16x16 pa;
#pragma unroll
    for (int e = 0; e < 16; ++e) {
      const int kk = kmap(e, kb);
      pa[e] = (kk < 16) ? (__bf16)Pl[wave][lm][kk] : (__bf16)0.0f;
    }
    // V B-frags from LDS, accumulate O (16x64 as four 16x16 tiles)
#pragma unroll
    for (int j = 0; j < 4; ++j) {
      bf16x16 vf;
#pragma unroll
      for (int e = 0; e < 16; ++e) {
        const int kk = kmap(e, kb);
        vf[e] = (kk < 16) ? (__bf16)Vl[wave][kk][j * 16 + lm] : (__bf16)0.0f;
      }
      o[j] = wmma_bf16(pa, vf, o[j]);
    }
  }

  // epilogue: normalize by l_i and write ctx
#pragma unroll
  for (int r = 0; r < 8; ++r) {
    const float inv = 1.0f / li[r];
    const int srow = qi * 16 + half * 8 + r;
    float* op = ctx + (size_t)(b * S + srow) * D + h * DK;
#pragma unroll
    for (int j = 0; j < 4; ++j) op[j * 16 + lm] = o[j][r] * inv;
  }
}

// ---------------------------------------------------------------------------
// out[row] = LayerNorm(A[row] + B[row]) * g + beta      (D = 1024, 256 thr)
// ---------------------------------------------------------------------------
__global__ __launch_bounds__(256) void add_layernorm_kernel(
    const float* __restrict__ A, const float* __restrict__ B,
    const float* __restrict__ g, const float* __restrict__ beta,
    float* __restrict__ out) {
  const int D = 1024;
  const int row = blockIdx.x;
  const float* pa = A + (size_t)row * D;
  const float* pb = B + (size_t)row * D;
  const int lane = threadIdx.x & 31;
  const int wave = threadIdx.x >> 5;

  float vals[4];
  float s = 0.0f;
#pragma unroll
  for (int i = 0; i < 4; ++i) {
    const int idx = threadIdx.x + 256 * i;
    const float v = pa[idx] + pb[idx];
    vals[i] = v;
    s += v;
  }
#pragma unroll
  for (int o = 16; o > 0; o >>= 1) s += __shfl_xor(s, o, 32);
  __shared__ float ws_sum[8];
  if (lane == 0) ws_sum[wave] = s;
  __syncthreads();
  float tot = 0.0f;
#pragma unroll
  for (int w = 0; w < 8; ++w) tot += ws_sum[w];
  const float mu = tot * (1.0f / 1024.0f);

  float sq = 0.0f;
#pragma unroll
  for (int i = 0; i < 4; ++i) { const float d = vals[i] - mu; sq += d * d; }
#pragma unroll
  for (int o = 16; o > 0; o >>= 1) sq += __shfl_xor(sq, o, 32);
  __shared__ float ws_sq[8];
  if (lane == 0) ws_sq[wave] = sq;
  __syncthreads();
  float totq = 0.0f;
#pragma unroll
  for (int w = 0; w < 8; ++w) totq += ws_sq[w];
  const float rstd = rsqrtf(totq * (1.0f / 1024.0f) + 1e-5f);

#pragma unroll
  for (int i = 0; i < 4; ++i) {
    const int idx = threadIdx.x + 256 * i;
    out[(size_t)row * D + idx] = (vals[i] - mu) * rstd * g[idx] + beta[idx];
  }
}

// ---------------------------------------------------------------------------
// Host-side orchestration.
// Inputs: x, mask, Wq, bq, Wk, bk, Wv, bv, Wo, bo, ln1_g, ln1_b, W1, b1,
//         W2, b2, ln2_g, ln2_b
// ---------------------------------------------------------------------------
extern "C" void kernel_launch(void* const* d_in, const int* in_sizes, int n_in,
                              void* d_out, int out_size, void* d_ws, size_t ws_size,
                              hipStream_t stream) {
  const float* x    = (const float*)d_in[0];
  const int*   mask = (const int*)d_in[1];
  const float* Wq = (const float*)d_in[2];   const float* bq = (const float*)d_in[3];
  const float* Wk = (const float*)d_in[4];   const float* bk = (const float*)d_in[5];
  const float* Wv = (const float*)d_in[6];   const float* bv = (const float*)d_in[7];
  const float* Wo = (const float*)d_in[8];   const float* bo = (const float*)d_in[9];
  const float* g1 = (const float*)d_in[10];  const float* be1 = (const float*)d_in[11];
  const float* W1 = (const float*)d_in[12];  const float* b1 = (const float*)d_in[13];
  const float* W2 = (const float*)d_in[14];  const float* b2 = (const float*)d_in[15];
  const float* g2 = (const float*)d_in[16];  const float* be2 = (const float*)d_in[17];
  float* out = (float*)d_out;

  const int BS = 4 * 2048;  // B*S rows
  float* ws = (float*)d_ws;
  float* q   = ws;                               // [BS,1024]
  float* kk  = ws + (size_t)1 * BS * 1024;       // [BS,1024]
  float* vv  = ws + (size_t)2 * BS * 1024;       // [BS,1024]
  float* ctx = ws + (size_t)3 * BS * 1024;       // [BS,1024]
  float* ff  = ws + (size_t)4 * BS * 1024;       // [BS,4096]
  float* attn_out = q;   // q dead after attention
  float* h        = kk;  // k dead after attention
  float* ff2      = vv;  // v dead after attention

  dim3 blk(256);
  // Q/K/V projections: grid (N/128, M/64)
  gemm_bias_kernel<0><<<dim3(8, 128), blk, 0, stream>>>(x, Wq, bq, q,  BS, 1024, 1024);
  gemm_bias_kernel<0><<<dim3(8, 128), blk, 0, stream>>>(x, Wk, bk, kk, BS, 1024, 1024);
  gemm_bias_kernel<0><<<dim3(8, 128), blk, 0, stream>>>(x, Wv, bv, vv, BS, 1024, 1024);
  // flash attention: 8192 (b,h,qtile) waves = 1024 blocks x 8 waves
  attention_kernel<<<dim3(1024), blk, 0, stream>>>(q, kk, vv, mask, ctx);
  // output projection + residual LN
  gemm_bias_kernel<0><<<dim3(8, 128), blk, 0, stream>>>(ctx, Wo, bo, attn_out, BS, 1024, 1024);
  add_layernorm_kernel<<<dim3(BS), blk, 0, stream>>>(x, attn_out, g1, be1, h);
  // FFN (GELU fused into first GEMM) + residual LN
  gemm_bias_kernel<1><<<dim3(32, 128), blk, 0, stream>>>(h, W1, b1, ff, BS, 4096, 1024);
  gemm_bias_kernel<0><<<dim3(8, 128), blk, 0, stream>>>(ff, W2, b2, ff2, BS, 1024, 4096);
  add_layernorm_kernel<<<dim3(BS), blk, 0, stream>>>(h, ff2, g2, be2, out);
}
